// MyLSTM_16183436771656
// MI455X (gfx1250) — matrix operand order repaired and checked by reading, not compiled, not run on hardware
//
#include <hip/hip_runtime.h>
#include <hip/hip_bf16.h>

#define DEV __device__ __forceinline__

typedef __attribute__((ext_vector_type(8)))  __bf16 v8bf;
typedef __attribute__((ext_vector_type(16))) __bf16 v16bf;
typedef __attribute__((ext_vector_type(8)))  float  v8f;
typedef __attribute__((ext_vector_type(4)))  unsigned int v4u;
typedef __attribute__((ext_vector_type(8)))  int v8i;
typedef __attribute__((ext_vector_type(4)))  int v4i;

constexpr int Hh   = 1024;    // HIDDEN
constexpr int Bt   = 128;     // batch
constexpr int Tt   = 128;     // timesteps
constexpr int INN  = 128;     // IN
constexpr int OUTN = 128;     // OUT
constexpr int NBLK = 128;     // 64 hidden slices x 2 batch halves
constexpr int PH   = 1040;    // W_hh LDS pitch (bf16): 512dw row + 2x4dw TDM pads = 520 dwords
constexpr int PI   = 136;     // W_ih LDS pitch (bf16): 64dw row + 4dw TDM pad = 68 dwords

// ---------------- scalar helpers ----------------
DEV unsigned short f2bf(float f) {           // fp32 -> bf16, round-to-nearest-even
  unsigned int u = __float_as_uint(f);
  u += 0x7FFFu + ((u >> 16) & 1u);
  return (unsigned short)(u >> 16);
}
DEV float sigm(float x)  { return 1.0f / (1.0f + __expf(-x)); }
DEV float tanh_(float x) { float e = __expf(2.0f * x); return (e - 1.0f) / (e + 1.0f); }

// ---------------- WMMA operand loaders (contiguous k = 16h + s, same order in A and B) ----------------
// 16-bit A matrix 16x32 (MxK):
//   lanes 0-15 : M=lane,    K = k0..k0+15
//   lanes 16-31: M=lane-16, K = k0+16..k0+31
DEV v16bf load_a_tile(const __bf16* base, int ld, int m0, int k0, int lane) {
  const int ln = lane & 15;
  const int hf = lane >> 4;
  const __bf16* p = base + (size_t)(m0 + ln) * ld + (k0 + hf * 16);
  v8bf lo = *(const v8bf*)(p);
  v8bf hi = *(const v8bf*)(p + 8);
  return __builtin_shufflevector(lo, hi, 0,1,2,3,4,5,6,7,8,9,10,11,12,13,14,15);
}
// 16-bit B matrix 32x16 (KxN), B[k][n] = W[n0+n][k] (weights row-major [N,K]):
//   lanes 0-15 : N=lane,    K = k0..k0+15 ; lanes 16-31: N=lane-16, K = k0+16..k0+31
DEV v16bf load_b_tile(const __bf16* base, int ld, int n0, int k0, int lane) {
  const int ln = lane & 15;
  const int hf = lane >> 4;
  const __bf16* p = base + (size_t)(n0 + ln) * ld + (k0 + hf * 16);
  v8bf lo = *(const v8bf*)(p);
  v8bf hi = *(const v8bf*)(p + 8);
  return __builtin_shufflevector(lo, hi, 0,1,2,3,4,5,6,7,8,9,10,11,12,13,14,15);
}
// B operand from the TDM-staged W_hh LDS image: a 16B pad is inserted after every
// 256 dwords (512 bf16), so element offsets >= 512 shift by 8 bf16.
DEV v16bf load_b_whh(const __bf16* sWhh, int row0, int k0, int lane) {
  const int ln = lane & 15;
  const int hf = lane >> 4;
  int ke = k0 + hf * 16;
  ke += (ke >= 512) ? 8 : 0;
  const __bf16* p = sWhh + (row0 + ln) * PH + ke;
  v8bf lo = *(const v8bf*)(p);
  v8bf hi = *(const v8bf*)(p + 8);
  return __builtin_shufflevector(lo, hi, 0,1,2,3,4,5,6,7,8,9,10,11,12,13,14,15);
}
DEV v8f wmma_bf16(v16bf a, v16bf b, v8f c) {
  return __builtin_amdgcn_wmma_f32_16x16x32_bf16(false, a, false, b, (short)0, c, false, false);
}

// ---------------- Tensor Data Mover: 3D tile (Z=gates, Y=rows, X=K) -> LDS ----------------
// D# per CDNA5 ISA sec. 8.3-8.6; data_size=2B; pad_enable inserts LDS pitch padding.
// amdgpu-toolchain (clang-23) 6-arg builtin form.
DEV void tdm_load_3d(unsigned lds_byte_off, unsigned long long gaddr,
                     unsigned tensor_d0, unsigned tile_d0, unsigned tile_d1, unsigned tile_d2,
                     unsigned long long d0_stride, unsigned long long d1_stride,
                     unsigned pad_interval, unsigned pad_amount) {
  v4u g0; v8i g1; v4i g2, g3; v8i g4;
  g0[0] = 1u;                                                   // count=1, user descriptor
  g0[1] = lds_byte_off;                                         // lds_addr
  g0[2] = (unsigned)(gaddr & 0xFFFFFFFFull);                    // global_addr[31:0]
  g0[3] = (unsigned)((gaddr >> 32) & 0x1FFFFFFull) | (2u << 30);// global_addr[56:32] | type=2
  g1[0] = (int)((1u << 16)                // data_size = 1 (2 bytes)
              | (1u << 20)                // pad_enable
              | (pad_interval << 22)      // pads every 2<<pad_interval dwords
              | (pad_amount << 25));      // pad_amount+1 dwords
  g1[1] = (int)((tensor_d0 & 0xFFFFu) << 16);                         // tensor_dim0[15:0]
  g1[2] = (int)(((tensor_d0 >> 16) & 0xFFFFu) | (tile_d1 << 16));     // dim0 hi | tensor_dim1 lo
  g1[3] = (int)((tile_d0 & 0xFFFFu) << 16);                           // tensor_dim1 hi=0 | tile_dim0
  g1[4] = (int)((tile_d1 & 0xFFFFu) | ((tile_d2 & 0xFFFFu) << 16));   // tile_dim1 | tile_dim2
  g1[5] = (int)(unsigned)(d0_stride & 0xFFFFFFFFull);                 // tensor_dim0_stride lo
  g1[6] = (int)(((d0_stride >> 32) & 0xFFFFull)
              | ((unsigned)(d1_stride & 0xFFFFull) << 16));           // d0 hi | d1_stride lo
  g1[7] = (int)(unsigned)((d1_stride >> 16) & 0xFFFFFFFFull);         // tensor_dim1_stride hi
  g2[0] = (int)tile_d2; g2[1] = 0; g2[2] = 0; g2[3] = 0;              // tensor_dim2
  g3[0] = 0; g3[1] = 0; g3[2] = 0; g3[3] = 0;
  g4[0] = 0; g4[1] = 0; g4[2] = 0; g4[3] = 0;                         // extra group: zero-filled
  g4[4] = 0; g4[5] = 0; g4[6] = 0; g4[7] = 0;
  __builtin_amdgcn_tensor_load_to_lds(g0, g1, g2, g3, g4, 0);
}

// ---------------- device-wide barrier (all NBLK blocks resident) ----------------
DEV void grid_barrier(int* cnt, int* gen) {
  __threadfence();                 // publish h stores device-wide
  __syncthreads();
  if (threadIdx.x == 0) {
    int g = __hip_atomic_load(gen, __ATOMIC_RELAXED, __HIP_MEMORY_SCOPE_AGENT);
    int v = __hip_atomic_fetch_add(cnt, 1, __ATOMIC_ACQ_REL, __HIP_MEMORY_SCOPE_AGENT);
    if (v == NBLK - 1) {
      __hip_atomic_store(cnt, 0, __ATOMIC_RELAXED, __HIP_MEMORY_SCOPE_AGENT);
      __hip_atomic_fetch_add(gen, 1, __ATOMIC_RELEASE, __HIP_MEMORY_SCOPE_AGENT);
    } else {
      while (__hip_atomic_load(gen, __ATOMIC_ACQUIRE, __HIP_MEMORY_SCOPE_AGENT) == g)
        __builtin_amdgcn_s_sleep(1);
    }
  }
  __syncthreads();
}

// ---------------- prep kernels ----------------
__global__ void cvt_f32_bf16(const float* __restrict__ s, unsigned short* __restrict__ d, int n) {
  for (int i = blockIdx.x * blockDim.x + threadIdx.x; i < n; i += gridDim.x * blockDim.x)
    d[i] = f2bf(s[i]);
}
__global__ void init_bar_kernel(int* p) { p[0] = 0; p[1] = 0; }

// ---------------- persistent LSTM kernel ----------------
// grid = 128 blocks x 128 threads (4 waves each).
// block b: jblk=b>>1 -> hidden cols j0=jblk*16 ; mh=b&1 -> batch half.
// wave w owns batch tile m0 = mh*64 + w*16, all 4 gate tiles, and the c-state tile.
__global__ __launch_bounds__(128) void lstm_persist(
    const unsigned short* __restrict__ xbf,
    const unsigned short* __restrict__ nbf,     // noises bf16 [T,B,IN]
    const unsigned short* __restrict__ Wehb,    // [H,OUT]
    const unsigned short* __restrict__ Wecb,    // [H,OUT]
    const unsigned short* __restrict__ Wihb,    // [4H,IN]
    const unsigned short* __restrict__ Whhb,    // [4H,H]
    const unsigned short* __restrict__ Woutb,   // [OUT,H]
    const float* __restrict__ b_eh, const float* __restrict__ b_ec,
    const float* __restrict__ b_ih, const float* __restrict__ b_hh,
    const float* __restrict__ b_out,
    unsigned short* __restrict__ hbuf,          // [2][B][H] bf16, double-buffered
    float* __restrict__ out,                    // [T][B][OUT]
    int* bar_cnt, int* bar_gen)
{
  extern __shared__ char smem[];
  __bf16* sWhh = (__bf16*)smem;          // [64][PH]  rows: gate*16 + local hidden col
  __bf16* sWih = sWhh + 64 * PH;         // [64][PI]

  const int tid  = threadIdx.x;
  const int wv   = tid >> 5;
  const int lane = tid & 31;
  const int ln   = lane & 15;
  const int jblk = blockIdx.x >> 1;
  const int mh   = blockIdx.x & 1;
  const int j0   = jblk * 16;            // hidden slice
  const int m0   = mh * 64 + wv * 16;    // batch tile

  // ---- TDM: stage this block's W_hh / W_ih slices into LDS (HBM read once) ----
  if (wv == 0) {
    const unsigned ldsWhh = (unsigned)(unsigned long long)(uintptr_t)sWhh;
    const unsigned ldsWih = (unsigned)(unsigned long long)(uintptr_t)sWih;
    // W_hh: tile 4(gates) x 16(rows) x 1024(K); row stride H, gate stride H*H elems.
    // Pads: 4 dwords per 256 dwords -> LDS pitch 520 dwords (PH=1040 bf16).
    tdm_load_3d(ldsWhh,
                (unsigned long long)(uintptr_t)Whhb + (unsigned long long)j0 * Hh * 2ull,
                Hh, Hh, 16, 4,
                (unsigned long long)Hh, (unsigned long long)Hh * Hh,
                7u /*256 dwords*/, 3u /*4 dwords*/);
    // W_ih: tile 4 x 16 x 128; pads: 4 dwords per 64 dwords -> pitch 68 dwords (PI=136 bf16).
    tdm_load_3d(ldsWih,
                (unsigned long long)(uintptr_t)Wihb + (unsigned long long)j0 * INN * 2ull,
                INN, INN, 16, 4,
                (unsigned long long)INN, (unsigned long long)Hh * INN,
                5u /*64 dwords*/, 3u /*4 dwords*/);
    __builtin_amdgcn_s_wait_tensorcnt((short)0);
  }
  __syncthreads();

  const int roff = (lane < 16) ? 0 : 8;  // C/D tile: VGPR e -> row m0+e / m0+8+e

  // ---- h0 = x@W_eh^T + b_eh ; c0 = x@W_ec^T + b_ec (c stays in registers) ----
  v8f ch = {0,0,0,0,0,0,0,0};
  {
    v8f ah = {0,0,0,0,0,0,0,0};
    v8f ac = {0,0,0,0,0,0,0,0};
    const __bf16* xb = (const __bf16*)xbf;
    for (int k = 0; k < INN; k += 32) {
      v16bf a = load_a_tile(xb, INN, m0, k, lane);
      ah = wmma_bf16(a, load_b_tile((const __bf16*)Wehb, OUTN, j0, k, lane), ah);
      ac = wmma_bf16(a, load_b_tile((const __bf16*)Wecb, OUTN, j0, k, lane), ac);
    }
    const float beh = b_eh[j0 + ln], bec = b_ec[j0 + ln];
#pragma unroll
    for (int e = 0; e < 8; ++e) {
      ch[e] = ac[e] + bec;
      hbuf[(size_t)(m0 + roff + e) * Hh + j0 + ln] = f2bf(ah[e] + beh);
    }
  }
  grid_barrier(bar_cnt, bar_gen);

  // per-lane gate biases (column depends only on ln)
  const float bi  = b_ih[0 * Hh + j0 + ln] + b_hh[0 * Hh + j0 + ln];
  const float bff = b_ih[1 * Hh + j0 + ln] + b_hh[1 * Hh + j0 + ln];
  const float bg  = b_ih[2 * Hh + j0 + ln] + b_hh[2 * Hh + j0 + ln];
  const float bo  = b_ih[3 * Hh + j0 + ln] + b_hh[3 * Hh + j0 + ln];

  // dx tile assignment: the 64 mh==0 blocks cover the 8x8 grid of 16x16 output tiles
  const int mdx0   = (jblk >> 3) * 16;
  const int ndx0   = (jblk & 7) * 16;
  const bool do_dx = (mh == 0) && (wv == (jblk & 3));

  for (int t = 0; t < Tt; ++t) {
    const __bf16* rd = (const __bf16*)(hbuf + (size_t)(t & 1) * Bt * Hh);
    unsigned short* wr = hbuf + (size_t)((t & 1) ^ 1) * Bt * Hh;

    v8f gi = {0,0,0,0,0,0,0,0}, gf = {0,0,0,0,0,0,0,0};
    v8f gg = {0,0,0,0,0,0,0,0}, go = {0,0,0,0,0,0,0,0};

    // input contribution: eps_t @ W_ih^T (K = 128, B operand from LDS)
    const __bf16* eps = (const __bf16*)(nbf + (size_t)t * Bt * INN);
    for (int k = 0; k < INN; k += 32) {
      v16bf a = load_a_tile(eps, INN, m0, k, lane);
      gi = wmma_bf16(a, load_b_tile(sWih, PI,  0, k, lane), gi);
      gf = wmma_bf16(a, load_b_tile(sWih, PI, 16, k, lane), gf);
      gg = wmma_bf16(a, load_b_tile(sWih, PI, 32, k, lane), gg);
      go = wmma_bf16(a, load_b_tile(sWih, PI, 48, k, lane), go);
    }
    // recurrent contribution: h @ W_hh^T (K = 1024, B operand from LDS)
    for (int k = 0; k < Hh; k += 32) {
      v16bf a = load_a_tile(rd, Hh, m0, k, lane);
      gi = wmma_bf16(a, load_b_whh(sWhh,  0, k, lane), gi);
      gf = wmma_bf16(a, load_b_whh(sWhh, 16, k, lane), gf);
      gg = wmma_bf16(a, load_b_whh(sWhh, 32, k, lane), gg);
      go = wmma_bf16(a, load_b_whh(sWhh, 48, k, lane), go);
    }

    // elementwise cell update, all in registers; publish h_new as bf16
#pragma unroll
    for (int e = 0; e < 8; ++e) {
      float iv = sigm(gi[e] + bi);
      float fv = sigm(gf[e] + bff);
      float gv = tanh_(gg[e] + bg);
      float ov = sigm(go[e] + bo);
      float cv = fv * ch[e] + iv * gv;
      ch[e] = cv;
      wr[(size_t)(m0 + roff + e) * Hh + j0 + ln] = f2bf(ov * tanh_(cv));
    }

    grid_barrier(bar_cnt, bar_gen);   // h_new visible device-wide

    // dx_t = tanh(h_new @ W_out^T + b_out): one 16x16 tile per designated wave
    if (do_dx) {
      v8f acc = {0,0,0,0,0,0,0,0};
      const __bf16* hn = (const __bf16*)wr;
      for (int k = 0; k < Hh; k += 32) {
        v16bf a = load_a_tile(hn, Hh, mdx0, k, lane);
        v16bf b = load_b_tile((const __bf16*)Woutb, Hh, ndx0, k, lane);
        acc = wmma_bf16(a, b, acc);
      }
      const float bb = b_out[ndx0 + ln];
      float* op = out + (size_t)t * Bt * OUTN;
#pragma unroll
      for (int e = 0; e < 8; ++e)
        op[(size_t)(mdx0 + roff + e) * OUTN + ndx0 + ln] = tanh_(acc[e] + bb);
    }
  }
}

// ---------------- host side ----------------
extern "C" void kernel_launch(void* const* d_in, const int* in_sizes, int n_in,
                              void* d_out, int out_size, void* d_ws, size_t ws_size,
                              hipStream_t stream) {
  const float* x      = (const float*)d_in[0];
  const float* noises = (const float*)d_in[1];
  const float* W_eh   = (const float*)d_in[2];
  const float* b_eh   = (const float*)d_in[3];
  const float* W_ec   = (const float*)d_in[4];
  const float* b_ec   = (const float*)d_in[5];
  const float* W_ih   = (const float*)d_in[6];
  const float* b_ih   = (const float*)d_in[7];
  const float* W_hh   = (const float*)d_in[8];
  const float* b_hh   = (const float*)d_in[9];
  const float* W_out  = (const float*)d_in[10];
  const float* b_out  = (const float*)d_in[11];
  float* out = (float*)d_out;

  // workspace carve-up (~14.6 MB total)
  char* ws = (char*)d_ws;
  int* bar = (int*)ws;                                       // 2 ints (256B slot)
  unsigned short* hbuf  = (unsigned short*)(ws + 256);       // 2*128*1024
  unsigned short* xbf   = hbuf  + 2 * Bt * Hh;               // 128*128
  unsigned short* nbf   = xbf   + Bt * OUTN;                 // 128*128*128
  unsigned short* Wehb  = nbf   + (size_t)Tt * Bt * INN;     // 1024*128
  unsigned short* Wecb  = Wehb  + Hh * OUTN;                 // 1024*128
  unsigned short* Wihb  = Wecb  + Hh * OUTN;                 // 4096*128
  unsigned short* Whhb  = Wihb  + 4 * Hh * INN;              // 4096*1024
  unsigned short* Woutb = Whhb  + (size_t)4 * Hh * Hh;       // 128*1024

  init_bar_kernel<<<1, 1, 0, stream>>>(bar);
  auto cvt = [&](const float* s, unsigned short* d, int n) {
    int blocks = (n + 255) / 256; if (blocks > 4096) blocks = 4096;
    cvt_f32_bf16<<<blocks, 256, 0, stream>>>(s, d, n);
  };
  cvt(x,      xbf,   Bt * OUTN);
  cvt(noises, nbf,   Tt * Bt * INN);
  cvt(W_eh,   Wehb,  Hh * OUTN);
  cvt(W_ec,   Wecb,  Hh * OUTN);
  cvt(W_ih,   Wihb,  4 * Hh * INN);
  cvt(W_hh,   Whhb,  4 * Hh * Hh);
  cvt(W_out,  Woutb, OUTN * Hh);

  const size_t smem = (size_t)(64 * PH + 64 * PI) * sizeof(unsigned short); // ~147 KB < 320 KB/WGP
  (void)hipFuncSetAttribute((const void*)lstm_persist,
                            hipFuncAttributeMaxDynamicSharedMemorySize, (int)smem);
  lstm_persist<<<NBLK, 128, smem, stream>>>(
      xbf, nbf, Wehb, Wecb, Wihb, Whhb, Woutb,
      b_eh, b_ec, b_ih, b_hh, b_out,
      hbuf, out, bar, bar + 1);
}
